// ClassAttentionTSSA_65738769433254
// MI455X (gfx1250) — matrix operand order repaired and checked by reading, not compiled, not run on hardware
//
#include <hip/hip_runtime.h>
#include <hip/hip_bf16.h>
#include <math.h>

// ---------------------------------------------------------------------------
// ClassAttentionTSSA on gfx1250 (MI455X): fused WMMA flash-attention.
// B=64, C=256, T=64, V=25 (pad->32), h=8, hd=32, TV=1600, key-block NB=64.
// ---------------------------------------------------------------------------

typedef _Float16 v16h  __attribute__((ext_vector_type(16)));
typedef _Float16 half8 __attribute__((ext_vector_type(8)));
typedef _Float16 half2v __attribute__((ext_vector_type(2)));
typedef float    v8f   __attribute__((ext_vector_type(8)));
typedef float    f32x4 __attribute__((ext_vector_type(4)));

#define B_   64
#define C_   256
#define V_   25
#define TV_  1600
#define H_   8
#define HD_  32
#define NB_  64
#define NITER_ (TV_ / NB_)
#define XP_PITCH 264   // 264*2B = 528B rows: 16B-aligned chunks, decorrelated banks

static __device__ __forceinline__ v8f wmma16x16x32(v16h a, v16h b, v8f c) {
  // (neg_a, A, neg_b, B, c_mod, C, reuse_a, reuse_b)
  return __builtin_amdgcn_wmma_f32_16x16x32_f16(false, a, false, b, (short)0, c,
                                                false, false);
}

static __device__ __forceinline__ v16h cat8(half8 lo, half8 hi) {
  return __builtin_shufflevector(lo, hi, 0, 1, 2, 3, 4, 5, 6, 7, 8, 9, 10, 11,
                                 12, 13, 14, 15);
}

// A-matrix fragment (16 x 32 f16) from row-major src[row][k] (pitch in halfs).
// lanes 0-15 hold row M=lane, K=k0..k0+15;
// lanes 16-31 hold row M=lane-16, K=k0+16..k0+31.
static __device__ __forceinline__ v16h load_a16(const _Float16* src, int pitch,
                                                int row0, int k0, int lane) {
  const int hs = (lane >> 4) & 1;
  const _Float16* r = src + (row0 + (lane & 15)) * pitch + k0 + hs * 16;
  half8 lo = *(const half8*)(r);
  half8 hi = *(const half8*)(r + 8);
  return cat8(lo, hi);
}

// Same A-fragment pattern, but from f32 source (weights), converted to f16.
static __device__ __forceinline__ v16h load_a32(const float* src, int pitch,
                                                int row0, int k0, int lane) {
  const int hs = (lane >> 4) & 1;
  const float* r = src + (row0 + (lane & 15)) * pitch + k0 + hs * 16;
  v16h o = {};
#pragma unroll
  for (int i = 0; i < 2; ++i) {
    f32x4 x = ((const f32x4*)r)[i];            // k0+hs*16 .. +7
    f32x4 y = ((const f32x4*)(r + 8))[i];      // k0+hs*16+8 .. +7
    o[4 * i + 0] = (_Float16)x.x;  o[4 * i + 1] = (_Float16)x.y;
    o[4 * i + 2] = (_Float16)x.z;  o[4 * i + 3] = (_Float16)x.w;
    o[8 + 4 * i + 0] = (_Float16)y.x;  o[8 + 4 * i + 1] = (_Float16)y.y;
    o[8 + 4 * i + 2] = (_Float16)y.z;  o[8 + 4 * i + 3] = (_Float16)y.w;
  }
  return o;
}

// B-matrix fragment (32 x 16 f16): element (k, n) at src[n*pitch + k].
// Lane L holds column n = L&15, 16 contiguous K starting at k0 + (L>=16)*16.
static __device__ __forceinline__ v16h load_b16(const _Float16* src, int pitch,
                                                int col0, int k0, int lane) {
  const int hs = (lane >> 4) & 1;
  const _Float16* p = src + (col0 + (lane & 15)) * pitch + k0 + hs * 16;
  half8 lo = *(const half8*)(p);
  half8 hi = *(const half8*)(p + 8);
  return cat8(lo, hi);
}

// B fragment from f32 weights (row-major src[n][k]), converted to f16 on load.
static __device__ __forceinline__ v16h load_b32(const float* src, int pitch,
                                                int col0, int k0, int lane) {
  const int hs = (lane >> 4) & 1;
  const float* p = src + (col0 + (lane & 15)) * pitch + k0 + hs * 16;
  v16h r = {};
#pragma unroll
  for (int i = 0; i < 4; ++i) {
    f32x4 x = ((const f32x4*)p)[i];
    r[4 * i + 0] = (_Float16)x.x;
    r[4 * i + 1] = (_Float16)x.y;
    r[4 * i + 2] = (_Float16)x.z;
    r[4 * i + 3] = (_Float16)x.w;
  }
  return r;
}

// Pack a D-tile accumulator (8 f32) to 8 consecutive f16 and store to LDS.
static __device__ __forceinline__ void store_half8(_Float16* dst, v8f acc) {
  half8 hv = {};
#pragma unroll
  for (int r = 0; r < 8; ++r) hv[r] = (_Float16)acc[r];
  *(half8*)dst = hv;
}

// ---------------------------------------------------------------------------
// Kernel 1: Q projection. qws[b][v][c'] = sum_c x_cls[b][c][v] * q_w[c'][c]
// padded rows v=25..31 are zero. Tiny (0.2 GFLOP) -> plain VALU.
// ---------------------------------------------------------------------------
__global__ void __launch_bounds__(256) qproj_kernel(
    const float* __restrict__ x_cls, const float* __restrict__ q_w,
    _Float16* __restrict__ qws) {
  const int idx = blockIdx.x * 256 + threadIdx.x;  // b*8192 + vrow*256 + c'
  const int c = idx & (C_ - 1);
  const int vrow = (idx >> 8) & 31;
  const int b = idx >> 13;
  float acc = 0.f;
  if (vrow < V_) {
    const float* xc = x_cls + b * C_ * V_ + vrow;  // stride V_ over channel dim
    const float* wr = q_w + c * C_;
#pragma unroll 8
    for (int k = 0; k < C_; ++k) acc += xc[k * V_] * wr[k];
  }
  qws[idx] = (_Float16)acc;
}

// ---------------------------------------------------------------------------
// Kernel 2: per-(batch, head) fused K/V projection + flash attention.
// Grid (64, 8), 256 threads = 8 waves.
// ---------------------------------------------------------------------------
__global__ void __launch_bounds__(256) attn_kernel(
    const float* __restrict__ x_patch, const float* __restrict__ k_w,
    const float* __restrict__ v_w, const float* __restrict__ temp,
    const _Float16* __restrict__ qws, _Float16* __restrict__ attn_ws) {
  __shared__ _Float16 xp[NB_ * XP_PITCH];   // xp[key][c]   (33 KB)
  __shared__ _Float16 kblk[NB_ * HD_];      // kblk[key][d] (4 KB)
  __shared__ _Float16 vblk[HD_ * NB_];      // vblk[d][key] (4 KB)
  __shared__ float    sblk[32 * NB_];       // scores       (8 KB)
  __shared__ _Float16 pblk[32 * NB_];       // probs        (4 KB)
  __shared__ float    segmax[32 * 8];       // per-row segment maxima (1 KB)
  __shared__ float    segsum[32 * 8];       // per-row segment sums   (1 KB)
  __shared__ float    mrow[32];             // broadcast running max
  __shared__ float    rescale[32];
  __shared__ float    linv[32];

  const int b = blockIdx.x, h = blockIdx.y;
  const int tid = threadIdx.x;
  const int w = tid >> 5, lane = tid & 31;
  const int hs = (lane >> 4) & 1;

  const float tscale = temp[h] * 0.17677669529663687f;  // temp[h] / sqrt(32)

  const int smt = w >> 2, snt = w & 3;        // S-tile per wave (2x4 tiles)
  const int kdmt = w >> 2, kknt = w & 3;      // K-proj: dims as M (2), keys as N (4)
  const int vkmt = w & 3, vdnt = w >> 2;      // V-proj: keys as M (4), dims as N (2)
  const int omt = (w >> 1) & 1, ont = w & 1;  // O-tile, waves 0..3 (2x2)
  const int srow = tid >> 3, sseg = tid & 7;  // softmax (row, 8-key segment)

  // Q fragment for this wave's S row-tile (rows 25..31 are zero-padded).
  const v16h aq = load_a16(qws + (size_t)b * 32 * C_, C_, smt * 16, h * HD_, lane);

  v8f acco = {};
  float m = -INFINITY, l = 0.f;

  const float* xpb = x_patch + (size_t)b * C_ * TV_;
  const float* kwh = k_w + h * HD_ * C_;
  const float* vwh = v_w + h * HD_ * C_;

  for (int it = 0; it < NITER_; ++it) {
    const int n0 = it * NB_;
    __syncthreads();
    // Stage xp tile (transpose x_patch[b][c][n] -> xp[key][c], f32 -> f16).
    // Two channel rows per thread so converts pack (v_cvt_pk_f16_f32) and
    // LDS stores are b32.
#pragma unroll
    for (int rep = 0; rep < 8; ++rep) {
      const int idx = rep * 256 + tid;
      const int c = (idx >> 4) * 2, j = (idx & 15) * 4;
      f32x4 xa = *(const f32x4*)(xpb + c * TV_ + n0 + j);
      f32x4 xb = *(const f32x4*)(xpb + (c + 1) * TV_ + n0 + j);
#pragma unroll
      for (int i = 0; i < 4; ++i) {
        half2v p;
        p.x = (_Float16)xa[i];
        p.y = (_Float16)xb[i];
        *(half2v*)(&xp[(j + i) * XP_PITCH + c]) = p;
      }
    }
    // Prefetch next key block while this one is being consumed.
    {
      const int n1 = (it + 1 < NITER_) ? (it + 1) * NB_ : 0;
      const float* nx = xpb + tid * TV_ + n1;  // tid == channel row
      __builtin_prefetch(nx, 0, 3);
      __builtin_prefetch(nx + 32, 0, 3);
    }
    __syncthreads();

    // K projection, transposed orientation: Ktile^T = k_w(16 dims x 32c) @ xp^T.
    // D-tile: M=dim, N=key -> 8 consecutive dims per lane -> packed b128 store.
    {
      v8f ck = {};
#pragma unroll
      for (int ks = 0; ks < 8; ++ks) {
        v16h ak = load_a32(kwh, C_, kdmt * 16, ks * 32, lane);  // hoisted
        v16h bx = load_b16(xp, XP_PITCH, kknt * 16, ks * 32, lane);
        ck = wmma16x16x32(ak, bx, ck);
      }
      const int key = kknt * 16 + (lane & 15);
      store_half8(&kblk[key * HD_ + kdmt * 16 + 8 * hs], ck);
    }
    // V projection, natural orientation: Vtile = xp(16 keys x 32c) @ v_w^T.
    // D-tile: M=key, N=dim -> 8 consecutive keys per lane -> packed b128 store.
    {
      v8f cv = {};
#pragma unroll
      for (int ks = 0; ks < 8; ++ks) {
        v16h ax = load_a16(xp, XP_PITCH, vkmt * 16, ks * 32, lane);
        v16h bv = load_b32(vwh, C_, vdnt * 16, ks * 32, lane);  // hoisted
        cv = wmma16x16x32(ax, bv, cv);
      }
      const int d = vdnt * 16 + (lane & 15);
      store_half8(&vblk[d * NB_ + vkmt * 16 + 8 * hs], cv);
    }
    __syncthreads();

    // S = Q @ K^T (one 16x16x32 WMMA per wave), scaled.
    {
      v16h bS = load_b16(kblk, HD_, snt * 16, 0, lane);
      v8f z = {};
      v8f s = wmma16x16x32(aq, bS, z);
#pragma unroll
      for (int r = 0; r < 8; ++r) {
        const int qr = smt * 16 + 8 * hs + r;
        sblk[qr * NB_ + snt * 16 + (lane & 15)] = s[r] * tscale;
      }
    }
    __syncthreads();

    // Online softmax, parallel over all 256 threads (8 segments per row).
    {  // Phase A: segment maxima
      const float* s8 = sblk + srow * NB_ + sseg * 8;
      float mx = s8[0];
#pragma unroll
      for (int j = 1; j < 8; ++j) mx = fmaxf(mx, s8[j]);
      segmax[srow * 8 + sseg] = mx;
    }
    __syncthreads();
    if (tid < 32) {  // Phase B: row owners fold maxima, publish mnew/rescale
      float mb = segmax[tid * 8];
#pragma unroll
      for (int s = 1; s < 8; ++s) mb = fmaxf(mb, segmax[tid * 8 + s]);
      const float mnew = fmaxf(m, mb);
      rescale[tid] = __expf(m - mnew);  // exp(-inf)=0 on first block
      mrow[tid] = mnew;
      m = mnew;
    }
    __syncthreads();
    {  // Phase C: exponentials, probs to LDS, segment sums
      const float mn = mrow[srow];
      const float* s8 = sblk + srow * NB_ + sseg * 8;
      _Float16* p8 = pblk + srow * NB_ + sseg * 8;
      float sum = 0.f;
#pragma unroll
      for (int j = 0; j < 8; ++j) {
        const float p = __expf(s8[j] - mn);
        p8[j] = (_Float16)p;
        sum += p;
      }
      segsum[srow * 8 + sseg] = sum;
    }
    __syncthreads();
    if (tid < 32) {  // Phase D: fold sums into running denominator
      float sum = segsum[tid * 8];
#pragma unroll
      for (int s = 1; s < 8; ++s) sum += segsum[tid * 8 + s];
      l = l * rescale[tid] + sum;
    }

    // O = O*rescale + P @ V  (waves 0..3, 2 WMMA k-steps over 64 keys).
    if (w < 4) {
#pragma unroll
      for (int r = 0; r < 8; ++r) acco[r] *= rescale[omt * 16 + 8 * hs + r];
#pragma unroll
      for (int ks = 0; ks < 2; ++ks) {
        v16h aP = load_a16(pblk, NB_, omt * 16, ks * 32, lane);
        v16h bV = load_b16(vblk, NB_, ont * 16, ks * 32, lane);
        acco = wmma16x16x32(aP, bV, acco);
      }
    }
  }

  __syncthreads();
  if (tid < 32) linv[tid] = 1.0f / l;
  __syncthreads();
  if (w < 4) {
    _Float16* dst = attn_ws + (size_t)b * 32 * C_ + h * HD_;
#pragma unroll
    for (int r = 0; r < 8; ++r) {
      const int qr = omt * 16 + 8 * hs + r;
      dst[qr * C_ + ont * 16 + (lane & 15)] = (_Float16)(acco[r] * linv[qr]);
    }
  }
}

// ---------------------------------------------------------------------------
// Kernel 3: out = attn @ proj_w.T + proj_b, stored transposed to (B,C,1,V).
// One block per batch, 8 waves; wave w owns output-channel tiles {2w, 2w+1}.
// ---------------------------------------------------------------------------
__global__ void __launch_bounds__(256) proj_kernel(
    const _Float16* __restrict__ attn_ws, const float* __restrict__ proj_w,
    const float* __restrict__ proj_b, float* __restrict__ out) {
  const int b = blockIdx.x;
  const int tid = threadIdx.x, w = tid >> 5, lane = tid & 31;
  const int hs = (lane >> 4) & 1;
  const _Float16* arows = attn_ws + (size_t)b * 32 * C_;

  v8f acc[2][2] = {};  // [m-tile][local n-tile]
#pragma unroll
  for (int ks = 0; ks < 8; ++ks) {
    v16h a0 = load_a16(arows, C_, 0, ks * 32, lane);
    v16h a1 = load_a16(arows, C_, 16, ks * 32, lane);
    v16h b0 = load_b32(proj_w, C_, (w * 2 + 0) * 16, ks * 32, lane);
    v16h b1 = load_b32(proj_w, C_, (w * 2 + 1) * 16, ks * 32, lane);
    acc[0][0] = wmma16x16x32(a0, b0, acc[0][0]);
    acc[0][1] = wmma16x16x32(a0, b1, acc[0][1]);
    acc[1][0] = wmma16x16x32(a1, b0, acc[1][0]);
    acc[1][1] = wmma16x16x32(a1, b1, acc[1][1]);
  }
#pragma unroll
  for (int ntl = 0; ntl < 2; ++ntl) {
    const int cc = (w * 2 + ntl) * 16 + (lane & 15);
    const float bias = proj_b[cc];
#pragma unroll
    for (int mt = 0; mt < 2; ++mt) {
#pragma unroll
      for (int r = 0; r < 8; ++r) {
        const int vr = mt * 16 + 8 * hs + r;
        if (vr < V_)
          out[(size_t)b * C_ * V_ + cc * V_ + vr] = acc[mt][ntl][r] + bias;
      }
    }
  }
}

// ---------------------------------------------------------------------------
extern "C" void kernel_launch(void* const* d_in, const int* in_sizes, int n_in,
                              void* d_out, int out_size, void* d_ws,
                              size_t ws_size, hipStream_t stream) {
  (void)in_sizes; (void)n_in; (void)out_size; (void)ws_size;
  const float* x_cls   = (const float*)d_in[0];
  const float* x_patch = (const float*)d_in[1];
  const float* q_w     = (const float*)d_in[2];
  const float* k_w     = (const float*)d_in[3];
  const float* v_w     = (const float*)d_in[4];
  const float* temp    = (const float*)d_in[5];
  const float* proj_w  = (const float*)d_in[6];
  const float* proj_b  = (const float*)d_in[7];
  float* out = (float*)d_out;

  _Float16* qws = (_Float16*)d_ws;                 // (B, 32, 256) f16 = 1 MB
  _Float16* aws = qws + (size_t)B_ * 32 * C_;      // (B, 32, 256) f16 = 1 MB

  qproj_kernel<<<(B_ * 32 * C_) / 256, 256, 0, stream>>>(x_cls, q_w, qws);
  attn_kernel<<<dim3(B_, H_), 256, 0, stream>>>(x_patch, k_w, v_w, temp, qws,
                                                aws);
  proj_kernel<<<dim3(B_), 256, 0, stream>>>(aws, proj_w, proj_b, out);
}
